// HomoGNN_74577812128299
// MI455X (gfx1250) — matrix operand order repaired and checked by reading, not compiled, not run on hardware
//
#include <hip/hip_runtime.h>
#include <hip/hip_bf16.h>

typedef __bf16 bf16;
typedef bf16  v16bf __attribute__((ext_vector_type(16)));
typedef bf16  v8bf  __attribute__((ext_vector_type(8)));
typedef float v8f   __attribute__((ext_vector_type(8)));

// Native CDNA5 f32 atomic add (no return -> STOREcnt tracked, no wait needed;
// device scope so the RMW is coherent across workgroups and resolves in L2).
__device__ __forceinline__ void global_fadd(float* p, float v) {
    asm volatile("global_atomic_add_f32 %0, %1, off scope:SCOPE_DEV"
                 :
                 : "v"(p), "v"(v)
                 : "memory");
}

// ---------------- degree / norm ----------------
__global__ void k_fill_deg(unsigned* __restrict__ deg, int n) {
    int i = blockIdx.x * blockDim.x + threadIdx.x;
    if (i < n) deg[i] = 1u;                       // self-loop
}

__global__ void k_count_deg(const int* __restrict__ dst, unsigned* __restrict__ deg, int e) {
    int i = blockIdx.x * blockDim.x + threadIdx.x;
    if (i < e) atomicAdd(&deg[dst[i]], 1u);
}

__global__ void k_dis(const unsigned* __restrict__ deg, float* __restrict__ dis, int n) {
    int i = blockIdx.x * blockDim.x + threadIdx.x;
    if (i < n) dis[i] = rsqrtf((float)deg[i]);    // deg >= 1 always
}

// ---------------- dtype conversion ----------------
__global__ void k_f32_to_bf16(const float* __restrict__ in, bf16* __restrict__ out, long long n) {
    long long i = (long long)blockIdx.x * blockDim.x + threadIdx.x;
    if (i < n) out[i] = (bf16)in[i];
}

__global__ void k_relu_to_bf16(const float* __restrict__ in, bf16* __restrict__ out, long long n) {
    long long i = (long long)blockIdx.x * blockDim.x + threadIdx.x;
    if (i < n) out[i] = (bf16)fmaxf(in[i], 0.0f);
}

// ---------------- weight repack into WMMA B-fragment order ----------------
// B tile is 32(K) x 16(N). Lane L, element e of v16bf holds B[(e<8 ? 8*(L>>4)+e : 16+8*(L>>4)+(e-8))][L&15].
// Packed layout: [kt][nt][lane][e] contiguous -> one 32B load per lane per tile.
__global__ void k_pack_w(const float* __restrict__ W, bf16* __restrict__ Bp, int K, int NOUT) {
    int idx = blockIdx.x * blockDim.x + threadIdx.x;
    int total = (K >> 5) * (NOUT >> 4) * 512;
    if (idx >= total) return;
    int e    = idx & 15;
    int lane = (idx >> 4) & 31;
    int tile = idx >> 9;
    int NT   = NOUT >> 4;
    int nt   = tile % NT;
    int kt   = tile / NT;
    int kr   = kt * 32 + ((e < 8) ? (8 * (lane >> 4) + e) : (16 + 8 * (lane >> 4) + (e - 8)));
    int nc   = nt * 16 + (lane & 15);
    Bp[idx]  = (bf16)W[(long long)kr * NOUT + nc];
}

// ---------------- WMMA GEMM: C[M x NOUT] = A[M x K](bf16) * Wpacked (+bias) ----------------
// One wave per block; block computes a 16-row x 64-col tile (4 wmma n-tiles).
template<int K, int NOUT>
__global__ __launch_bounds__(32)
void k_gemm_wmma(const bf16* __restrict__ A, const bf16* __restrict__ Bp,
                 float* __restrict__ C, const float* __restrict__ bias, int M) {
    const int lane = threadIdx.x;      // 0..31
    const int col  = lane & 15;
    const int hi   = lane >> 4;        // 0 or 1
    const int m0   = blockIdx.x * 16;
    constexpr int NT = NOUT / 16;
    const int nt0  = blockIdx.y * 4;

    // A fragment row for this lane (clamped for safety at the ragged edge)
    int arow_i = m0 + col;
    if (arow_i >= M) arow_i = M - 1;
    const bf16* arow = A + (long long)arow_i * K;

    v8f acc[4] = {};

    for (int k0 = 0; k0 < K; k0 += 32) {
        // A: lane holds row (lane&15); K elems = {k0+8*hi .. +8} ++ {k0+16+8*hi .. +8}
        v8bf a_lo = *(const v8bf*)(arow + k0 + 8 * hi);
        v8bf a_hi = *(const v8bf*)(arow + k0 + 16 + 8 * hi);
        v16bf a;
#pragma unroll
        for (int e = 0; e < 8; ++e) { a[e] = a_lo[e]; a[e + 8] = a_hi[e]; }

        const int kt = k0 >> 5;
#pragma unroll
        for (int t = 0; t < 4; ++t) {
            const bf16* bp = Bp + ((long long)(kt * NT + nt0 + t) << 9) + (lane << 4);
            v16bf b = *(const v16bf*)bp;
            acc[t] = __builtin_amdgcn_wmma_f32_16x16x32_bf16(
                false, a, false, b, (short)0, acc[t], false, false);
        }
    }

    // C layout: VGPR r -> row (r + 8*hi), col = lane&15 within tile
#pragma unroll
    for (int t = 0; t < 4; ++t) {
        const int  ccol = (nt0 + t) * 16 + col;
        const float bv  = bias ? bias[ccol] : 0.0f;
#pragma unroll
        for (int r = 0; r < 8; ++r) {
            const int crow = m0 + 8 * hi + r;
            if (crow < M)
                C[(long long)crow * NOUT + ccol] = acc[t][r] + bv;
        }
    }
}

// ---------------- aggregation ----------------
// out = bias + self_loop(norm = dis^2) ; full init (no stale data, no atomics)
template<int C>
__global__ void k_init_agg(const float* __restrict__ h, const float* __restrict__ dis,
                           const float* __restrict__ b, float* __restrict__ out, long long n) {
    long long i = (long long)blockIdx.x * blockDim.x + threadIdx.x;
    if (i >= n) return;
    constexpr int SH = (C == 256) ? 8 : 7;
    int node = (int)(i >> SH);
    int c    = (int)(i & (C - 1));
    float d  = dis[node];
    out[i] = b[c] + h[i] * (d * d);
}

// one thread per (edge, 4-channel chunk): float4 gather + 4 native f32 atomic scatters
template<int C>
__global__ void k_edge_agg(const int* __restrict__ src, const int* __restrict__ dst,
                           const float* __restrict__ dis, const float* __restrict__ h,
                           float* __restrict__ out, long long total) {
    long long idx = (long long)blockIdx.x * blockDim.x + threadIdx.x;
    if (idx >= total) return;
    constexpr int Q = C / 4;
    int q = (int)(idx % Q);
    long long e = idx / Q;
    int s = src[e];
    int d = dst[e];
    float w = dis[s] * dis[d];
    const float4 v = *(const float4*)(h + (long long)s * C + q * 4);
    float* o = out + (long long)d * C + q * 4;
    global_fadd(o + 0, v.x * w);
    global_fadd(o + 1, v.y * w);
    global_fadd(o + 2, v.z * w);
    global_fadd(o + 3, v.w * w);
}

// ---------------- launcher ----------------
static inline int gblk(long long n, int b) { return (int)((n + b - 1) / b); }

extern "C" void kernel_launch(void* const* d_in, const int* in_sizes, int n_in,
                              void* d_out, int out_size, void* d_ws, size_t ws_size,
                              hipStream_t stream) {
    const float* x  = (const float*)d_in[0];
    const int*   ei = (const int*)  d_in[1];   // [2, E] : src row then dst row
    const float* W1 = (const float*)d_in[2];
    const float* b1 = (const float*)d_in[3];
    const float* W2 = (const float*)d_in[4];
    const float* b2 = (const float*)d_in[5];
    const float* Wl = (const float*)d_in[6];
    const float* bl = (const float*)d_in[7];
    float* out = (float*)d_out;

    const int N = in_sizes[0] / 256;   // 50000
    const int E = in_sizes[1] / 2;     // 800000
    const int* src = ei;
    const int* dst = ei + E;

    // workspace carve (256B aligned chunks)
    char* p = (char*)d_ws;
    auto carve = [&](size_t bytes) -> void* {
        void* r = (void*)p;
        p += (bytes + 255) & ~((size_t)255);
        return r;
    };
    unsigned* deg = (unsigned*)carve((size_t)N * 4);
    float*    dis = (float*)   carve((size_t)N * 4);
    bf16*    bufA = (bf16*)    carve((size_t)N * 256 * 2);  // bf16 activations (reused each layer)
    float*   hlin = (float*)   carve((size_t)N * 256 * 4);  // GEMM output
    float*   hagg = (float*)   carve((size_t)N * 256 * 4);  // aggregation output
    bf16*      p1 = (bf16*)    carve((size_t)8 * 16 * 512 * 2);  // W1 packed (256->256)
    bf16*      p2 = (bf16*)    carve((size_t)8 *  8 * 512 * 2);  // W2 packed (256->128)
    bf16*      p3 = (bf16*)    carve((size_t)4 *  8 * 512 * 2);  // Wl packed (128->128)

    const int B = 256;
    const int MT = (N + 15) / 16;       // 3125

    // degrees + deg^-1/2
    k_fill_deg <<<gblk(N, B), B, 0, stream>>>(deg, N);
    k_count_deg<<<gblk(E, B), B, 0, stream>>>(dst, deg, E);
    k_dis      <<<gblk(N, B), B, 0, stream>>>(deg, dis, N);

    // pack weights into fragment order
    k_pack_w<<<gblk(8 * 16 * 512, B), B, 0, stream>>>(W1, p1, 256, 256);
    k_pack_w<<<gblk(8 *  8 * 512, B), B, 0, stream>>>(W2, p2, 256, 128);
    k_pack_w<<<gblk(4 *  8 * 512, B), B, 0, stream>>>(Wl, p3, 128, 128);

    // ---- layer 1: h = relu( prop(x @ W1) + b1 ) ----
    k_f32_to_bf16<<<gblk((long long)N * 256, B), B, 0, stream>>>(x, bufA, (long long)N * 256);
    k_gemm_wmma<256, 256><<<dim3(MT, 4), 32, 0, stream>>>(bufA, p1, hlin, nullptr, N);
    k_init_agg<256><<<gblk((long long)N * 256, B), B, 0, stream>>>(hlin, dis, b1, hagg, (long long)N * 256);
    {
        long long total = (long long)E * 64;   // E * (256/4)
        k_edge_agg<256><<<gblk(total, B), B, 0, stream>>>(src, dst, dis, hlin, hagg, total);
    }
    k_relu_to_bf16<<<gblk((long long)N * 256, B), B, 0, stream>>>(hagg, bufA, (long long)N * 256);

    // ---- layer 2: h = relu( prop(h @ W2) + b2 ) ----
    k_gemm_wmma<256, 128><<<dim3(MT, 2), 32, 0, stream>>>(bufA, p2, hlin, nullptr, N);
    k_init_agg<128><<<gblk((long long)N * 128, B), B, 0, stream>>>(hlin, dis, b2, hagg, (long long)N * 128);
    {
        long long total = (long long)E * 32;   // E * (128/4)
        k_edge_agg<128><<<gblk(total, B), B, 0, stream>>>(src, dst, dis, hlin, hagg, total);
    }
    k_relu_to_bf16<<<gblk((long long)N * 128, B), B, 0, stream>>>(hagg, bufA, (long long)N * 128);

    // ---- final linear: out = h @ Wl + bl ----
    k_gemm_wmma<128, 128><<<dim3(MT, 2), 32, 0, stream>>>(bufA, p3, out, bl, N);
}
